// Context32KScaling_4681514352741
// MI455X (gfx1250) — matrix operand order repaired and checked by reading, not compiled, not run on hardware
//
#include <hip/hip_runtime.h>
#include <math.h>

// Problem shape (fixed by the reference): B=8, T=2048, H=256, one head.
#define B_ 8
#define T_ 2048
#define H_ 256
#define BM 64          // query rows per workgroup (4 waves x 16 rows)
#define BN 32          // kv rows per inner tile (one WMMA K=32 for the PV gemm)
#define NWAVES 4
#define NKV (T_ / BN)
#define SCALE_ 0.0625f       // 1/sqrt(256)
#define SLOPE_ 0.00390625f   // 2^-8 alibi slope

typedef _Float16 half_t;
typedef __attribute__((ext_vector_type(16))) _Float16 v16h;
typedef __attribute__((ext_vector_type(8)))  _Float16 v8h;
typedef __attribute__((ext_vector_type(8)))  float    v8f;
typedef __attribute__((ext_vector_type(4)))  unsigned int u32x4;
typedef __attribute__((ext_vector_type(8)))  int i32x8;
typedef __attribute__((ext_vector_type(4)))  int i32x4;

union AF { v16h v; v8h h[2]; };   // A-matrix fragment: 16 halves/lane = 8 VGPRs

#if __has_builtin(__builtin_amdgcn_tensor_load_to_lds)
#define HAVE_TDM 1
#else
#define HAVE_TDM 0
#endif

// ---------------------------------------------------------------------------
// Pass 1a: RoPE(q,k) + f32->f16 into workspace. One block per (b,t) row.
// ---------------------------------------------------------------------------
__global__ void rope_cvt_kernel(const float* __restrict__ q,
                                const float* __restrict__ k,
                                half_t* __restrict__ qe,
                                half_t* __restrict__ ke) {
  const int row = blockIdx.x;              // b*T + t
  const int t   = row & (T_ - 1);
  const int d   = threadIdx.x;             // 0..127
  const size_t base = (size_t)row * H_;

  const float inv_freq = __expf(-(2.0f * (float)d / (float)H_) * __logf(10000.0f));
  const float ang = (float)t * inv_freq;
  const float c = __cosf(ang), s = __sinf(ang);

  const float q1 = q[base + d], q2 = q[base + d + H_ / 2];
  qe[base + d]          = (half_t)(q1 * c - q2 * s);
  qe[base + d + H_ / 2] = (half_t)(q2 * c + q1 * s);

  const float k1 = k[base + d], k2 = k[base + d + H_ / 2];
  ke[base + d]          = (half_t)(k1 * c - k2 * s);
  ke[base + d + H_ / 2] = (half_t)(k2 * c + k1 * s);
}

// ---------------------------------------------------------------------------
// Pass 1b: V (f32, [T][H]) -> V^T (f16, [H][T]) so PV B-operands are
// row-contiguous and the tile can be fetched by a plain 2D TDM load.
// ---------------------------------------------------------------------------
__global__ void transpose_v_kernel(const float* __restrict__ v,
                                   half_t* __restrict__ vt) {
  __shared__ half_t tile[32][33];
  const int b  = blockIdx.z;
  const int t0 = blockIdx.x * 32, h0 = blockIdx.y * 32;
  const int x  = threadIdx.x;              // 0..31
  for (int y = threadIdx.y; y < 32; y += 8)
    tile[y][x] = (half_t)v[(size_t)b * T_ * H_ + (size_t)(t0 + y) * H_ + h0 + x];
  __syncthreads();
  for (int y = threadIdx.y; y < 32; y += 8)
    vt[(size_t)b * H_ * T_ + (size_t)(h0 + y) * T_ + t0 + x] = tile[x][y];
}

// ---------------------------------------------------------------------------
// TDM helper: 2D f16 tile (tile_d0 x tile_d1) global -> LDS, one wave issue.
// D# group0: count|lds_addr|global_addr|type=2. group1: data_size=2B, dims.
// This toolchain exposes the 6-arg builtin:
//   (u32x4 g0, i32x8 g1, i32x4 g2, i32x4 g3, i32x8 pad, i32 cpol)
// ---------------------------------------------------------------------------
#if HAVE_TDM
__device__ __forceinline__ unsigned lds_off(const void* p) {
  return (unsigned)(uintptr_t)(const __attribute__((address_space(3))) void*)p;
}

__device__ __forceinline__ void tdm_load_2d_f16(unsigned lds_byte,
                                                const half_t* gptr,
                                                unsigned tensor_d0, unsigned tensor_d1,
                                                unsigned stride0,
                                                unsigned tile_d0, unsigned tile_d1) {
  const unsigned long long ga = (unsigned long long)(uintptr_t)gptr;
  u32x4 g0;
  g0[0] = 1u;                                              // count=1 descriptor
  g0[1] = lds_byte;                                        // LDS byte address
  g0[2] = (unsigned)ga;                                    // global addr [31:0]
  g0[3] = (unsigned)((ga >> 32) & 0x01FFFFFFu) | (2u << 30);   // [56:32] | type=2
  i32x8 g1;
  g1[0] = (int)(1u << 16);                                 // data_size = 2 bytes
  g1[1] = (int)((tensor_d0 & 0xFFFFu) << 16);              // tensor_dim0[15:0]
  g1[2] = (int)((tensor_d0 >> 16) | ((tensor_d1 & 0xFFFFu) << 16));
  g1[3] = (int)((tensor_d1 >> 16) | (tile_d0 << 16));      // tile_dim0
  g1[4] = (int)tile_d1;                                    // tile_dim1 (tile_dim2=0)
  g1[5] = (int)stride0;                                    // tensor_dim0_stride
  g1[6] = 0;
  g1[7] = 0;
  i32x4 z4; z4[0] = 0; z4[1] = 0; z4[2] = 0; z4[3] = 0;
  i32x8 z8; z8[0] = 0; z8[1] = 0; z8[2] = 0; z8[3] = 0;
  z8[4] = 0; z8[5] = 0; z8[6] = 0; z8[7] = 0;
  __builtin_amdgcn_tensor_load_to_lds(g0, g1, z4, z4, z8, 0);
}
#endif

// ---------------------------------------------------------------------------
// Pass 2: flash attention. 4 waves/block, each wave owns 16 query rows.
// TDM double-buffers K and V^T tiles (DMA overlaps WMMA); Q staged once.
// ---------------------------------------------------------------------------
__global__ void __launch_bounds__(128, 1)
flash_attn_kernel(const half_t* __restrict__ qe,
                  const half_t* __restrict__ ke,
                  const half_t* __restrict__ vt,
                  float* __restrict__ out) {
  __shared__ __align__(32) half_t lds_q[BM * H_];          // 32 KB, staged once
  __shared__ __align__(32) half_t lds_k[2][BN * H_];       // 2 x 16 KB
  __shared__ __align__(32) half_t lds_vt[2][H_ * BN];      // 2 x 16 KB
  __shared__ __align__(32) half_t lds_p[NWAVES * 16 * BN]; // 4 KB P scratch

  const int b     = blockIdx.y;
  const int tid   = threadIdx.x;
  const int wave  = tid >> 5;
  const int lane  = tid & 31;
  const int lhalf = lane >> 4;
  const int l16   = lane & 15;

  const size_t batch_off = (size_t)b * T_ * H_;
  const size_t vtb_off   = (size_t)b * H_ * T_;
  const int    m0        = blockIdx.x * BM + wave * 16;

#if HAVE_TDM
  if (wave == 0) {   // one wave drives the DMA engine
    tdm_load_2d_f16(lds_off(lds_q), qe + batch_off + (size_t)blockIdx.x * BM * H_,
                    H_, T_, H_, H_, BM);
    tdm_load_2d_f16(lds_off(lds_k[0]), ke + batch_off, H_, T_, H_, H_, BN);
    tdm_load_2d_f16(lds_off(lds_vt[0]), vt + vtb_off, T_, H_, T_, BN, H_);
  }
#else
  {
    const half_t* qb = qe + batch_off + (size_t)blockIdx.x * BM * H_;
#pragma unroll
    for (int tv = 0; tv < (BM * H_ / 8) / 128; ++tv) {
      const int vi = tid + tv * 128, row = vi >> 5, c8 = vi & 31;
      *(v8h*)&lds_q[row * H_ + c8 * 8] = *(const v8h*)(qb + (size_t)row * H_ + c8 * 8);
    }
  }
#endif

  v8f o[H_ / 16] = {};
  float m_run[8], l_run[8];
#pragma unroll
  for (int r = 0; r < 8; ++r) { m_run[r] = -3.0e38f; l_run[r] = 0.0f; }
  half_t* pbuf = lds_p + wave * (16 * BN);

  for (int it = 0; it < NKV; ++it) {
    const int kv0 = it * BN;
#if HAVE_TDM
    const int cur = it & 1;
    if (wave == 0) __builtin_amdgcn_s_wait_tensorcnt(0);   // tile `it` landed
    __syncthreads();                                       // ...visible to all
    if (wave == 0 && it + 1 < NKV) {                       // DMA next tile now;
      const int nxt = (it + 1) & 1;                        // overlaps WMMA below
      const int kvn = kv0 + BN;
      tdm_load_2d_f16(lds_off(lds_k[nxt]), ke + batch_off + (size_t)kvn * H_,
                      H_, T_, H_, H_, BN);
      tdm_load_2d_f16(lds_off(lds_vt[nxt]), vt + vtb_off + kvn,
                      T_, H_, T_, BN, H_);
    }
#else
    const int cur = 0;
    {
      const half_t* kb  = ke + batch_off + (size_t)kv0 * H_;
      const half_t* vb  = vt + vtb_off + kv0;
#pragma unroll
      for (int tv = 0; tv < (BN * H_ / 8) / 128; ++tv) {
        const int vi = tid + tv * 128, row = vi >> 5, c8 = vi & 31;
        *(v8h*)&lds_k[0][row * H_ + c8 * 8] = *(const v8h*)(kb + (size_t)row * H_ + c8 * 8);
      }
#pragma unroll
      for (int tv = 0; tv < (H_ * BN / 8) / 128; ++tv) {
        const int vi = tid + tv * 128, row = vi >> 2, c8 = vi & 3;
        *(v8h*)&lds_vt[0][row * BN + c8 * 8] = *(const v8h*)(vb + (size_t)row * T_ + c8 * 8);
      }
    }
    __syncthreads();
#endif

    // ---- S = Q K^T : two 16x16 f32 tiles over BN=32 kv columns ----
    v8f s0 = {}, s1 = {};
#pragma unroll
    for (int kc = 0; kc < H_ / 32; ++kc) {
      AF qa;
      const half_t* qp = &lds_q[(wave * 16 + l16) * H_ + kc * 32 + lhalf * 16];
      qa.h[0] = *(const v8h*)(qp);
      qa.h[1] = *(const v8h*)(qp + 8);
      v16h b0 = *(const v16h*)(&lds_k[cur][(0 * 16 + l16) * H_ + kc * 32 + lhalf * 16]);
      s0 = __builtin_amdgcn_wmma_f32_16x16x32_f16(false, qa.v, false, b0,
                                                  (short)0, s0, false, false);
      v16h b1 = *(const v16h*)(&lds_k[cur][(1 * 16 + l16) * H_ + kc * 32 + lhalf * 16]);
      s1 = __builtin_amdgcn_wmma_f32_16x16x32_f16(false, qa.v, false, b1,
                                                  (short)0, s1, false, false);
    }

    // ---- scale + alibi + online softmax (rows live in 16-lane groups) ----
    float alpha[8];
#pragma unroll
    for (int r = 0; r < 8; ++r) {
      const float qrow = (float)(m0 + r + lhalf * 8);
      const float col0 = (float)(kv0 + l16);
      float v0 = s0[r] * SCALE_ + SLOPE_ * (col0 - qrow);
      float v1 = s1[r] * SCALE_ + SLOPE_ * (col0 + 16.0f - qrow);
      float mx = fmaxf(v0, v1);
#pragma unroll
      for (int off = 8; off >= 1; off >>= 1)
        mx = fmaxf(mx, __shfl_xor(mx, off, 32));
      const float mnew = fmaxf(m_run[r], mx);
      const float a = __expf(m_run[r] - mnew);
      m_run[r] = mnew;
      const float p0 = __expf(v0 - mnew);
      const float p1 = __expf(v1 - mnew);
      s0[r] = p0; s1[r] = p1;
      float rs = p0 + p1;
#pragma unroll
      for (int off = 8; off >= 1; off >>= 1)
        rs += __shfl_xor(rs, off, 32);
      l_run[r] = l_run[r] * a + rs;
      alpha[r] = a;
    }
#pragma unroll
    for (int n = 0; n < H_ / 16; ++n)
#pragma unroll
      for (int r = 0; r < 8; ++r)
        o[n][r] *= alpha[r];

    // ---- P: f32 D-layout -> f16 A-layout via wave-private LDS round trip ----
#pragma unroll
    for (int r = 0; r < 8; ++r) {
      const int m = r + lhalf * 8;
      pbuf[m * BN + l16]      = (half_t)s0[r];
      pbuf[m * BN + 16 + l16] = (half_t)s1[r];
    }
    AF pf;
    {
      const half_t* pr = pbuf + (size_t)l16 * BN + lhalf * 16;
      pf.h[0] = *(const v8h*)(pr);
      pf.h[1] = *(const v8h*)(pr + 8);
    }

    // ---- O += P * V  (V^T rows in LDS give contiguous B columns) ----
#pragma unroll
    for (int n = 0; n < H_ / 16; ++n) {
      v16h bv = *(const v16h*)(&lds_vt[cur][(n * 16 + l16) * BN + lhalf * 16]);
      o[n] = __builtin_amdgcn_wmma_f32_16x16x32_f16(false, pf.v, false, bv,
                                                    (short)0, o[n], false, false);
    }
#if !HAVE_TDM
    __syncthreads();
#endif
  }

  // ---- epilogue: normalize by row sums and store fp32 ----
  float inv_l[8];
#pragma unroll
  for (int r = 0; r < 8; ++r) inv_l[r] = 1.0f / l_run[r];
  float* obase = out + batch_off;
#pragma unroll
  for (int n = 0; n < H_ / 16; ++n) {
#pragma unroll
    for (int r = 0; r < 8; ++r) {
      const int m = m0 + r + lhalf * 8;
      obase[(size_t)m * H_ + n * 16 + l16] = o[n][r] * inv_l[r];
    }
  }
}

// ---------------------------------------------------------------------------
extern "C" void kernel_launch(void* const* d_in, const int* in_sizes, int n_in,
                              void* d_out, int out_size, void* d_ws, size_t ws_size,
                              hipStream_t stream) {
  const float* q = (const float*)d_in[0];
  const float* k = (const float*)d_in[1];
  const float* v = (const float*)d_in[2];
  float* out = (float*)d_out;

  const size_t elems = (size_t)B_ * T_ * H_;
  if (ws_size < 3 * elems * sizeof(half_t)) return;   // 25.2 MB needed
  half_t* qe = (half_t*)d_ws;
  half_t* ke = qe + elems;
  half_t* vt = ke + elems;                            // V^T, [B][H][T]

  rope_cvt_kernel<<<dim3(B_ * T_), 128, 0, stream>>>(q, k, qe, ke);
  transpose_v_kernel<<<dim3(T_ / 32, H_ / 32, B_), dim3(32, 8), 0, stream>>>(v, vt);
  flash_attn_kernel<<<dim3(T_ / BM, B_), 128, 0, stream>>>(qe, ke, vt, out);
}
